// Cindex_32289564131755
// MI455X (gfx1250) — compile-verified, matrix operand order checked by reading, not run on hardware
//
#include <hip/hip_runtime.h>
#include <hip/hip_bf16.h>

typedef int v8i __attribute__((ext_vector_type(8)));

// ---------------------------------------------------------------------------
// Stage 0: zero the two u32 partial accumulators in workspace.
// ---------------------------------------------------------------------------
__global__ void cidx_zero(unsigned* partials) {
    partials[0] = 0u;
    partials[1] = 0u;
}

// ---------------------------------------------------------------------------
// Stage 1: full-matrix pair counting with wave32 ballot masks.
// Each lane owns one i (32 consecutive i per wave); j is wave-uniform so the
// j-side data comes in via scalar loads. One v_cmp per comparison covers 32
// pairs; all boolean combining is SALU mask logic; s_bcnt reduces 32 pairs
// per instruction. Grid: (N/256) i-blocks x 64 j-chunks.
// ---------------------------------------------------------------------------
__global__ __launch_bounds__(256) void cidx_pairs(const float* __restrict__ y,
                                                  const float* __restrict__ yh,
                                                  const int* __restrict__ st,
                                                  unsigned* __restrict__ partials,
                                                  int jchunk) {
    const int i = blockIdx.x * 256 + threadIdx.x;
    const float yi  = y[i];
    const float yhi = yh[i];
    // Per-wave mask of status bits for this wave's 32 i's (loop invariant).
    const unsigned si_mask = __builtin_amdgcn_ballot_w32(st[i] == 1);

    const int j0 = blockIdx.y * jchunk;
    unsigned cc = 0u, tp = 0u;

    #pragma unroll 4
    for (int j = j0; j < j0 + jchunk; ++j) {
        // j is uniform across the wave -> scalar loads (K$ / L2 resident).
        const float yj  = y[j];
        const float yhj = yh[j];
        const unsigned sj_mask = 0u - (unsigned)(st[j] == 1);  // 0 or ~0

        const unsigned G  = __builtin_amdgcn_ballot_w32(yi  >= yj);   // ge
        const unsigned Lm = __builtin_amdgcn_ballot_w32(yi  <= yj);   // le
        const unsigned Gh = __builtin_amdgcn_ballot_w32(yhi >= yhj);  // ge_hat
        const unsigned Lh = __builtin_amdgcn_ballot_w32(yhi <= yhj);  // le_hat

        const unsigned t1 = G  & sj_mask;   // ge & st_j
        const unsigned t2 = Lm & si_mask;   // le & st_i
        cc += (unsigned)__builtin_popcount((t1 & Gh) | (t2 & Lh));
        tp += (unsigned)__builtin_popcount(t1 | t2);
    }

    // Block reduction: 8 wave-uniform partials -> 1 atomic per counter.
    __shared__ unsigned scc[8];
    __shared__ unsigned stp[8];
    const int lane = threadIdx.x & 31;
    const int wv   = threadIdx.x >> 5;
    if (lane == 0) { scc[wv] = cc; stp[wv] = tp; }
    __syncthreads();
    if (threadIdx.x == 0) {
        unsigned a = 0u, b = 0u;
        #pragma unroll
        for (int k = 0; k < 8; ++k) { a += scc[k]; b += stp[k]; }
        atomicAdd(&partials[0], a);
        atomicAdd(&partials[1], b);
    }
}

// ---------------------------------------------------------------------------
// Stage 2 (single wave of 32): S = sum(status) via V_WMMA_I32_16X16X64_IU8
// with B = all-ones (D accumulates row sums exactly in i32; since B is all
// ones, the total equals the sum of every byte placed in A, independent of
// the exact lane/byte layout). Then emit (cc - S) / (tp - S) with the same
// int->f32 rounding as the reference.
// EXEC must be all ones before the WMMA loop: 32 threads, no divergence.
// ---------------------------------------------------------------------------
__global__ __launch_bounds__(32) void cidx_finalize(const int* __restrict__ st,
                                                    const unsigned* __restrict__ partials,
                                                    float* __restrict__ out,
                                                    int n) {
    const int lane = threadIdx.x;      // 0..31
    const int half = lane >> 4;        // 0: K-low bytes, 1: K-high bytes
    const int m    = lane & 15;        // A-matrix row

    v8i acc = {};
    v8i bones;
    #pragma unroll
    for (int r = 0; r < 8; ++r) bones[r] = 0x01010101;  // B = 64x16 of ones

    const int tiles = n >> 10;  // 1024 status values per 16x64 IU8 tile
    for (int t = 0; t < tiles; ++t) {
        const int base = (t << 10) + m * 64;
        v8i a;
        #pragma unroll
        for (int v = 0; v < 8; ++v) {
            // ISA 8-bit A layout: VGPR v holds 4 consecutive K bytes.
            const int k0 = (v >> 1) * 16 + half * 8 + (v & 1) * 4;
            const unsigned b0 = (unsigned)st[base + k0 + 0];
            const unsigned b1 = (unsigned)st[base + k0 + 1];
            const unsigned b2 = (unsigned)st[base + k0 + 2];
            const unsigned b3 = (unsigned)st[base + k0 + 3];
            a[v] = (int)(b0 | (b1 << 8) | (b2 << 16) | (b3 << 24));
        }
        // (sgn_a, A, sgn_b, B, C, reuse_a, reuse_b) -> v_wmma_i32_16x16x64_iu8
        acc = __builtin_amdgcn_wmma_i32_16x16x64_iu8(false, a, false, bones,
                                                     acc, false, false);
    }

    // S = sum over rows of D[m,0]: lane 0 holds M=0..7 in acc[0..7],
    // lane 16 holds M=8..15.
    unsigned S = 0u;
    #pragma unroll
    for (int r = 0; r < 8; ++r) {
        S += (unsigned)__builtin_amdgcn_readlane(acc[r], 0);
        S += (unsigned)__builtin_amdgcn_readlane(acc[r], 16);
    }

    if (lane == 0) {
        const unsigned ccAll = partials[0];
        const unsigned tpAll = partials[1];
        // Strict-upper-triangle counts (exact integers, both even diffs / 2).
        const float c = (float)((ccAll - S) >> 1);
        const float tot = (float)((tpAll - S) >> 1);
        out[0] = c / tot;
    }
}

// ---------------------------------------------------------------------------
extern "C" void kernel_launch(void* const* d_in, const int* in_sizes, int n_in,
                              void* d_out, int out_size, void* d_ws, size_t ws_size,
                              hipStream_t stream) {
    const float* y  = (const float*)d_in[0];
    const float* yh = (const float*)d_in[1];
    const int*   st = (const int*)d_in[2];
    float* out = (float*)d_out;
    unsigned* partials = (unsigned*)d_ws;

    const int N = in_sizes[0];            // 16384 (multiple of 1024)
    const int JCHUNKS = 64;
    const int jchunk = N / JCHUNKS;       // 256

    cidx_zero<<<1, 1, 0, stream>>>(partials);

    dim3 grid(N / 256, JCHUNKS);
    cidx_pairs<<<grid, 256, 0, stream>>>(y, yh, st, partials, jchunk);

    cidx_finalize<<<1, 32, 0, stream>>>(st, partials, out, N);
}
